// DLRM_Net_498216206942
// MI455X (gfx1250) — compile-verified, matrix operand order checked by reading, not run on hardware
//
#include <hip/hip_runtime.h>
#include <math.h>

#define BATCH   16384
#define NTAB    26
#define VOCABSZ 100000
#define EMB_D   64
#define NROWS   27                // 1 dense row + 26 embedding rows
#define TROW    (NROWS * EMB_D)   // 1728 bf16 per sample
#define NPAIR   351               // 27*26/2 lower-triangular pairs
#define RDIM    416               // 64 + 351 padded to multiple of 32

typedef __attribute__((ext_vector_type(16))) __bf16 v16bf;
typedef __attribute__((ext_vector_type(8)))  float  v8f;
typedef __attribute__((ext_vector_type(4)))  int    v4i;
typedef __attribute__((ext_vector_type(2)))  float  f32x2;
typedef unsigned short u16;
typedef unsigned int   u32;

union FragBF { v16bf v; int4 q[2]; };

// gfx1250 async global->LDS copy path (ASYNCcnt-tracked), if the toolchain
// exposes the builtins; otherwise fall back to load + ds_store staging.
#if defined(__has_builtin)
#  if __has_builtin(__builtin_amdgcn_global_load_async_to_lds_b128) && \
      __has_builtin(__builtin_amdgcn_s_wait_asynccnt)
#    define USE_ASYNC_COPY 1
#  endif
#endif
#ifndef USE_ASYNC_COPY
#  define USE_ASYNC_COPY 0
#endif

#if USE_ASYNC_COPY
typedef __attribute__((address_space(1))) v4i* v4i_gptr;
typedef __attribute__((address_space(3))) v4i* v4i_lptr;
#endif

__device__ __forceinline__ u16 f2bf(float f) {
    u32 u = __float_as_uint(f);
    u += 0x7FFFu + ((u >> 16) & 1u);      // round-to-nearest-even
    return (u16)(u >> 16);
}
__device__ __forceinline__ float bf2f(u16 h) {
    return __uint_as_float(((u32)h) << 16);
}

__device__ __forceinline__ void copy16B_to_lds(const u16* g, u16* l) {
#if USE_ASYNC_COPY
    __builtin_amdgcn_global_load_async_to_lds_b128((v4i_gptr)g, (v4i_lptr)l, 0, 0);
#else
    *reinterpret_cast<int4*>(l) = *reinterpret_cast<const int4*>(g);
#endif
}

// ---------------------------------------------------------------------------
// f32 [M x Kin] -> bf16 [M x Kout] with zero K-padding (also used for dense_x)
// ---------------------------------------------------------------------------
__global__ __launch_bounds__(256) void cvt_pad(const float* __restrict__ src,
                                               u16* __restrict__ dst,
                                               int M, int Kin, int Kout) {
    int idx = blockIdx.x * blockDim.x + threadIdx.x;
    if (idx >= M * Kout) return;
    int m = idx / Kout, k = idx - m * Kout;
    dst[idx] = (k < Kin) ? f2bf(src[(size_t)m * Kin + k]) : (u16)0;
}

// ---------------------------------------------------------------------------
// Embedding gather-sum: one wave per (table, sample).
// 32 lanes x float2 covers one 64-f32 row (256B coalesced). Random touch-once
// traffic -> non-temporal loads. Sum L=4 rows, write bf16 row into Tmat.
// ---------------------------------------------------------------------------
__global__ __launch_bounds__(256) void emb_gather(const float* __restrict__ emb,
                                                  const int* __restrict__ lS_i,
                                                  u16* __restrict__ T) {
    int gw   = (blockIdx.x * blockDim.x + threadIdx.x) >> 5;
    int lane = threadIdx.x & 31;
    if (gw >= NTAB * BATCH) return;
    int t = gw / BATCH;
    int b = gw - t * BATCH;
    const int*   ip   = lS_i + ((size_t)t * BATCH + b) * 4;
    const float* base = emb + (size_t)t * VOCABSZ * EMB_D;
    float s0 = 0.f, s1 = 0.f;
#pragma unroll
    for (int l = 0; l < 4; ++l) {
        int idx = ip[l];
        f32x2 v = __builtin_nontemporal_load(
            reinterpret_cast<const f32x2*>(base + (size_t)idx * EMB_D + lane * 2));
        s0 += v.x; s1 += v.y;
    }
    u32 packed = (u32)f2bf(s0) | ((u32)f2bf(s1) << 16);
    *reinterpret_cast<u32*>(T + (size_t)b * TROW + (t + 1) * EMB_D + lane * 2) = packed;
}

// ---------------------------------------------------------------------------
// LDS-staged, double-buffered bf16 WMMA GEMM:
//   out[b, m] = relu( sum_k A[b,k]*W[m,k] + bias[m] )
// Block = 8 waves = 128(batch) x 64(features) output tile.
// Per K-step(32): stage A(128x32) + W(64x32) bf16 tiles into LDS with
// coalesced 16B chunks (async global->LDS when available). All fragments for
// the step are pre-loaded (ds_load_b128 clause), then the four
// v_wmma_f32_16x16x32_bf16 issue back-to-back.
// Fragment layouts per CDNA5 ISA 7.12.2 (16-bit A 16x32; C/D f32 16x16).
// K multiple of 32 (zero-padded), M multiple of 64, BATCH multiple of 128.
// ---------------------------------------------------------------------------
__global__ __launch_bounds__(256) void gemm_bf16_bias_relu(
    const u16* __restrict__ A, int lda,
    const u16* __restrict__ W, int ldw,
    const float* __restrict__ bias,
    u16* __restrict__ out, int ldo,
    int M, int K)
{
    __shared__ u16 sA[2][128][32];      // 2 x 8KB
    __shared__ u16 sW[2][64][32];       // 2 x 4KB

    const int rowBlocks = BATCH / 128;  // 128
    const int blockRow  = blockIdx.x % rowBlocks;
    const int nBase     = (blockIdx.x / rowBlocks) * 64;
    const int rowBase0  = blockRow * 128;

    const int tid  = threadIdx.x;
    const int wave = tid >> 5;
    const int lane = tid & 31;
    const int lo16 = lane & 15;
    const int khi  = (lane >> 4) << 3;  // 0 or 8

    auto stage = [&](int buf, int kk) {
        // A tile: 128 rows x 64B = 512 x 16B chunks, 2 per thread
#pragma unroll
        for (int j = 0; j < 2; ++j) {
            int c   = tid + 256 * j;
            int row = c >> 2;
            int co  = (c & 3) * 8;
            copy16B_to_lds(A + (size_t)(rowBase0 + row) * lda + kk + co,
                           &sA[buf][row][co]);
        }
        // W tile: 64 rows x 64B = 256 x 16B chunks, 1 per thread
        {
            int row = tid >> 2;
            int co  = (tid & 3) * 8;
            copy16B_to_lds(W + (size_t)(nBase + row) * ldw + kk + co,
                           &sW[buf][row][co]);
        }
    };

    v8f acc[4];
#pragma unroll
    for (int t = 0; t < 4; ++t)
#pragma unroll
        for (int e = 0; e < 8; ++e) acc[t][e] = 0.0f;

    const int nSteps = K >> 5;
    stage(0, 0);
    for (int s = 0; s < nSteps; ++s) {
        const int cur = s & 1;
#if USE_ASYNC_COPY
        __builtin_amdgcn_s_wait_asynccnt(0);
#endif
        __syncthreads();                          // buffer `cur` ready
        if (s + 1 < nSteps) stage(cur ^ 1, (s + 1) * 32);

        // Pre-load all fragments for this step so the ds_loads clause up and
        // the 4 wmma issue back-to-back.
        FragBF fa, fb[4];
        const u16* sa = &sA[cur][(wave << 4) + lo16][khi];
        fa.q[0] = *reinterpret_cast<const int4*>(sa);
        fa.q[1] = *reinterpret_cast<const int4*>(sa + 16);
#pragma unroll
        for (int t = 0; t < 4; ++t) {
            const u16* sw = &sW[cur][(t << 4) + lo16][khi];
            fb[t].q[0] = *reinterpret_cast<const int4*>(sw);
            fb[t].q[1] = *reinterpret_cast<const int4*>(sw + 16);
        }
#pragma unroll
        for (int t = 0; t < 4; ++t) {
            acc[t] = __builtin_amdgcn_wmma_f32_16x16x32_bf16(
                false, fa.v, false, fb[t].v, (short)0, acc[t], false, false);
        }
        __syncthreads();                          // reads of `cur` done
    }

    const int rowBase = rowBase0 + (wave << 4);
#pragma unroll
    for (int t = 0; t < 4; ++t) {
        int   col = nBase + (t << 4) + lo16;
        float bv  = bias[col];
#pragma unroll
        for (int i = 0; i < 8; ++i) {
            int   row = rowBase + i + ((lane >> 4) << 3);
            float v   = acc[t][i] + bv;
            v = v > 0.0f ? v : 0.0f;              // all 5 fused layers use ReLU
            out[(size_t)row * ldo + col] = f2bf(v);
        }
    }
}

// ---------------------------------------------------------------------------
// Feature interaction: one wave per sample. Stage the 27x64 bf16 Tmat tile in
// LDS (per-wave region; DS ops are in-order within a wave so no barrier),
// compute 351 lower-tri dot products in f32, emit R = [x(64) | pairs | 0].
// ---------------------------------------------------------------------------
__global__ __launch_bounds__(256) void interact(const u16* __restrict__ T,
                                                u16* __restrict__ R) {
    __shared__ u16 sT[8][TROW];               // 8 waves * 3456B = 27.6 KB
    int wave = threadIdx.x >> 5, lane = threadIdx.x & 31;
    int b = blockIdx.x * 8 + wave;

    const u32* s32 = reinterpret_cast<const u32*>(T + (size_t)b * TROW);
    u32*       d32 = reinterpret_cast<u32*>(sT[wave]);
    for (int i = lane; i < TROW / 2; i += 32) d32[i] = s32[i];

    u16* Rb = R + (size_t)b * RDIM;
    // copy dense x (Tmat row 0) into R[0..63]
    reinterpret_cast<u32*>(Rb)[lane] = d32[lane];

    const u16* base = sT[wave];
    for (int p = lane; p < NPAIR; p += 32) {
        int i = (int)((1.0f + sqrtf(8.0f * (float)p + 1.0f)) * 0.5f);
        if (i * (i - 1) / 2 > p) --i;
        if ((i + 1) * i / 2 <= p) ++i;
        int j = p - i * (i - 1) / 2;
        const u32* ri = reinterpret_cast<const u32*>(base + i * EMB_D);
        const u32* rj = reinterpret_cast<const u32*>(base + j * EMB_D);
        float acc = 0.f;
#pragma unroll
        for (int d = 0; d < EMB_D / 2; ++d) {
            u32 a = ri[d], c = rj[d];
            acc += bf2f((u16)a)         * bf2f((u16)c);
            acc += bf2f((u16)(a >> 16)) * bf2f((u16)(c >> 16));
        }
        Rb[64 + p] = f2bf(acc);
    }
    if (lane == 0) Rb[RDIM - 1] = 0;          // zero K-pad column 415
}

// ---------------------------------------------------------------------------
// Final 256->1 layer + sigmoid: one wave per sample, shfl_xor reduction.
// ---------------------------------------------------------------------------
__global__ __launch_bounds__(256) void top_final(const u16* __restrict__ G1,
                                                 const float* __restrict__ W2,
                                                 const float* __restrict__ b2,
                                                 float* __restrict__ out) {
    int gw   = (blockIdx.x * blockDim.x + threadIdx.x) >> 5;
    int lane = threadIdx.x & 31;
    if (gw >= BATCH) return;
    const u16* h = G1 + (size_t)gw * 256;
    float acc = 0.f;
#pragma unroll
    for (int k = lane; k < 256; k += 32) acc += bf2f(h[k]) * W2[k];
#pragma unroll
    for (int off = 16; off > 0; off >>= 1) acc += __shfl_xor(acc, off, 32);
    if (lane == 0) out[gw] = 1.0f / (1.0f + __expf(-(acc + b2[0])));
}

// ---------------------------------------------------------------------------
// Host-side orchestration. Input order = setup_inputs() dict order:
// 0 dense_x, 1 lS_o(unused), 2 lS_i, 3 emb,
// 4 bot_W0, 5 bot_b0, 6 top_W0, 7 top_b0,
// 8 bot_W1, 9 bot_b1, 10 top_W1, 11 top_b1,
// 12 bot_W2, 13 bot_b2, 14 top_W2, 15 top_b2
// ---------------------------------------------------------------------------
extern "C" void kernel_launch(void* const* d_in, const int* in_sizes, int n_in,
                              void* d_out, int out_size, void* d_ws, size_t ws_size,
                              hipStream_t stream) {
    (void)in_sizes; (void)n_in; (void)out_size; (void)ws_size;

    const float* dense_x = (const float*)d_in[0];
    const int*   lS_i    = (const int*)d_in[2];
    const float* emb     = (const float*)d_in[3];
    const float* bW0 = (const float*)d_in[4];
    const float* bb0 = (const float*)d_in[5];
    const float* tW0 = (const float*)d_in[6];
    const float* tb0 = (const float*)d_in[7];
    const float* bW1 = (const float*)d_in[8];
    const float* bb1 = (const float*)d_in[9];
    const float* tW1 = (const float*)d_in[10];
    const float* tb1 = (const float*)d_in[11];
    const float* bW2 = (const float*)d_in[12];
    const float* bb2 = (const float*)d_in[13];
    const float* tW2 = (const float*)d_in[14];
    const float* tb2 = (const float*)d_in[15];
    float* out = (float*)d_out;

    char*  ws  = (char*)d_ws;
    size_t off = 0;
    auto alloc = [&](size_t bytes) -> char* {
        char* p = ws + off;
        off = (off + bytes + 255) & ~(size_t)255;
        return p;
    };
    u16* wB0 = (u16*)alloc((size_t)512 * 32  * 2);
    u16* wB1 = (u16*)alloc((size_t)256 * 512 * 2);
    u16* wB2 = (u16*)alloc((size_t)64  * 256 * 2);
    u16* wT0 = (u16*)alloc((size_t)512 * RDIM * 2);
    u16* wT1 = (u16*)alloc((size_t)256 * 512 * 2);
    u16* X0  = (u16*)alloc((size_t)BATCH * 32   * 2);
    u16* H0  = (u16*)alloc((size_t)BATCH * 512  * 2);   // reused for top layer 0 out
    u16* H1  = (u16*)alloc((size_t)BATCH * 256  * 2);   // reused for top layer 1 out
    u16* T   = (u16*)alloc((size_t)BATCH * TROW * 2);
    u16* R   = (u16*)alloc((size_t)BATCH * RDIM * 2);

    auto cvt = [&](const float* s, u16* d, int M, int Kin, int Kout) {
        int n = M * Kout;
        cvt_pad<<<(n + 255) / 256, 256, 0, stream>>>(s, d, M, Kin, Kout);
    };
    cvt(bW0, wB0, 512, 13, 32);
    cvt(bW1, wB1, 256, 512, 512);
    cvt(bW2, wB2, 64, 256, 256);
    cvt(tW0, wT0, 512, 415, RDIM);
    cvt(tW1, wT1, 256, 512, 512);
    cvt(dense_x, X0, BATCH, 13, 32);

    emb_gather<<<(NTAB * BATCH) / 8, 256, 0, stream>>>(emb, lS_i, T);

    auto gemm = [&](const u16* A, int lda, const u16* W, int ldw,
                    const float* bias, u16* o, int ldo, int M, int K) {
        int blocks = (BATCH / 128) * (M / 64);
        gemm_bf16_bias_relu<<<blocks, 256, 0, stream>>>(A, lda, W, ldw, bias, o, ldo, M, K);
    };
    // bottom MLP (layer 2 writes straight into Tmat row 0 via ldo = TROW)
    gemm(X0, 32,  wB0, 32,  bb0, H0, 512, 512, 32);
    gemm(H0, 512, wB1, 512, bb1, H1, 256, 256, 512);
    gemm(H1, 256, wB2, 256, bb2, T,  TROW, 64, 256);

    interact<<<BATCH / 8, 256, 0, stream>>>(T, R);

    // top MLP
    gemm(R,  RDIM, wT0, RDIM, tb0, H0, 512, 512, RDIM);
    gemm(H0, 512,  wT1, 512,  tb1, H1, 256, 256, 512);
    top_final<<<BATCH / 8, 256, 0, stream>>>(H1, tW2, tb2, out);
}
